// GAHT_59785944760981
// MI455X (gfx1250) — compile-verified, matrix operand order checked by reading, not run on hardware
//
#include <hip/hip_runtime.h>

// ---------------------------------------------------------------------------
// Types for CDNA5 WMMA (wave32): v_wmma_f32_16x16x32_bf16
// ---------------------------------------------------------------------------
typedef __bf16 bf16_t;
typedef bf16_t v16bf __attribute__((ext_vector_type(16)));
typedef bf16_t v8bf  __attribute__((ext_vector_type(8)));
typedef float  v8f   __attribute__((ext_vector_type(8)));

#define WMMA_BF16(a, b, c) \
  __builtin_amdgcn_wmma_f32_16x16x32_bf16(false, (a), false, (b), (short)0, (c), false, false)

// Async L2->LDS copy, 16B per lane, tracked by ASYNCcnt (CDNA5 §10 async ops).
// ldsoff: LDS byte address (low 32 bits of the generic shared-aperture address,
// per the flat-address truncation rule). gaddr: per-lane global address.
#define ASYNC_LD_B128(ldsoff, gaddr)                                   \
  asm volatile("global_load_async_to_lds_b128 %0, %1, off"             \
               :: "v"(ldsoff), "v"(gaddr) : "memory")
#define WAIT_ASYNCCNT0() asm volatile("s_wait_asynccnt 0x0" ::: "memory")

// Fragment loader: 16x32 bf16 tile, row-major source with element stride `ld`.
// Lane l<16 covers row l with kb = 0; lanes 16..31 cover the same rows with
// kb = 8. Two contiguous 16B loads per lane. Serves both A (rows = M) and B
// (rows = N of the pre-transposed [N][Kpad] weight), per §7.12.2 striping.
__device__ __forceinline__ v16bf load_frag16(const bf16_t* __restrict__ base,
                                             int ld, int lane) {
  const int r  = lane & 15;
  const int kb = (lane >> 4) << 3;  // 0 or 8
  const bf16_t* p = base + (size_t)r * ld + kb;
  v8bf lo = *(const v8bf*)(p);
  v8bf hi = *(const v8bf*)(p + 16);
  v16bf out;
#pragma unroll
  for (int i = 0; i < 8; ++i) { out[i] = lo[i]; out[i + 8] = hi[i]; }
  return out;
}

__device__ __forceinline__ float silu_f(float v) {
  return v / (1.0f + __expf(-v));
}

__device__ __forceinline__ unsigned lds_off_u32(const void* p) {
  return (unsigned)(uintptr_t)p;  // LDS aperture: addr[31:0] is the LDS byte address
}

// ---------------------------------------------------------------------------
// Weight convert + transpose: fp32 [K][Nsrc] -> bf16 [Npad][Kpad], zero padded
// ---------------------------------------------------------------------------
__global__ void wconv_kernel(const float* __restrict__ in, bf16_t* __restrict__ out,
                             int K, int Nsrc, int Npad, int Kpad) {
  int i = blockIdx.x * blockDim.x + threadIdx.x;
  if (i >= Npad * Kpad) return;
  int n = i / Kpad, k = i % Kpad;
  float v = (k < K && n < Nsrc) ? in[(size_t)k * Nsrc + n] : 0.0f;
  out[i] = (bf16_t)v;
}

// ---------------------------------------------------------------------------
// Edge distances
// ---------------------------------------------------------------------------
__global__ void dist_kernel(const float* __restrict__ pos, const int* __restrict__ src,
                            const int* __restrict__ dst, float* __restrict__ dist, int E) {
  int e = blockIdx.x * blockDim.x + threadIdx.x;
  if (e >= E) return;
  int s = src[e], d = dst[e];
  float dx = pos[(size_t)d * 3 + 0] - pos[(size_t)s * 3 + 0];
  float dy = pos[(size_t)d * 3 + 1] - pos[(size_t)s * 3 + 1];
  float dz = pos[(size_t)d * 3 + 2] - pos[(size_t)s * 3 + 2];
  dist[e] = sqrtf(dx * dx + dy * dy + dz * dz);
}

// ---------------------------------------------------------------------------
// Embedding: h = x @ emb_w + emb_b  (K=9, scalar; writes f32 + bf16)
// ---------------------------------------------------------------------------
__global__ void embed_kernel(const float* __restrict__ x, const float* __restrict__ w,
                             const float* __restrict__ b, float* __restrict__ h,
                             bf16_t* __restrict__ hb, int N) {
  int i = blockIdx.x * blockDim.x + threadIdx.x;
  if (i >= N * 128) return;
  int n = i >> 7, c = i & 127;
  float acc = b[c];
#pragma unroll
  for (int f = 0; f < 9; ++f) acc += x[(size_t)n * 9 + f] * w[f * 128 + c];
  h[i] = acc;
  hb[i] = (bf16_t)acc;
}

// ---------------------------------------------------------------------------
// Edge message kernel: per wave, 16 edges.
//   ef = [h[dst] | h[src] | dist | pad] (16 x 288 bf16 in LDS),
//        gathered with GLOBAL_LOAD_ASYNC_TO_LDS_B128 (ASYNCcnt path)
//   m1 = silu(ef @ w1 + b1)   (WMMA, 9 K-chunks)
//   m2 = silu(m1 @ w2 + b2)   (WMMA, 4 K-chunks)
//   aggr[dst] += m2           (global f32 atomics, L2-resident)
// ---------------------------------------------------------------------------
__global__ void __launch_bounds__(128)
edge_kernel(const bf16_t* __restrict__ hb, const float* __restrict__ dist,
            const int* __restrict__ srcI, const int* __restrict__ dstI,
            const bf16_t* __restrict__ w1t, const float* __restrict__ b1,
            const bf16_t* __restrict__ w2t, const float* __restrict__ b2,
            float* __restrict__ aggr, int E) {
  __shared__ bf16_t ef[4][16][288];   // 36 KB
  __shared__ bf16_t m1[4][16][128];   // 16 KB
  __shared__ int    dsts[4][16];
  const int wave = threadIdx.x >> 5, lane = threadIdx.x & 31;
  int e0 = (blockIdx.x * 4 + wave) * 16;
  if (e0 + 16 > E) e0 = E - 16;  // tail clamp (E is a multiple of 64 here)

  // ---- stage edge features into LDS: async L2->LDS, no VGPR round trip ----
  // lanes 0..15: 16B each of h[dst] row (256B); lanes 16..31: h[src] row.
  const int half = lane >> 4;           // 0 = dst half, 1 = src half
  const int sub  = lane & 15;           // 16B chunk within the row
#pragma unroll 4
  for (int e = 0; e < 16; ++e) {
    const int ed = e0 + e;
    const int di = dstI[ed], si = srcI[ed];
    if (lane == 0) dsts[wave][e] = di;
    const bf16_t* grow = hb + (size_t)(half ? si : di) * 128 + sub * 8;
    const unsigned loff = lds_off_u32(&ef[wave][e][half * 128 + sub * 8]);
    ASYNC_LD_B128(loff, grow);
  }
  if (lane < 16) ef[wave][lane][256] = (bf16_t)dist[e0 + lane];
#pragma unroll
  for (int e = 0; e < 16; ++e)
    if (lane < 31) ef[wave][e][257 + lane] = (bf16_t)0.0f;
  WAIT_ASYNCCNT0();
  __syncthreads();

  // ---- GEMM 1: [16 x 288] @ [288 x 128] ----
  const int rbase = (lane >> 4) << 3;
#pragma unroll 1
  for (int n0 = 0; n0 < 128; n0 += 16) {
    v8f acc = {};
    const bf16_t* Bb = w1t + (size_t)n0 * 288;
#pragma unroll
    for (int kc = 0; kc < 9; ++kc) {
      __builtin_prefetch(Bb + (kc + 1) * 32, 0, 1);
      v16bf a = load_frag16(&ef[wave][0][kc * 32], 288, lane);
      v16bf b = load_frag16(Bb + kc * 32, 288, lane);
      acc = WMMA_BF16(a, b, acc);
    }
    const int col = n0 + (lane & 15);
    const float bias = b1[col];
#pragma unroll
    for (int r = 0; r < 8; ++r) {
      float v = silu_f(acc[r] + bias);
      m1[wave][rbase + r][col] = (bf16_t)v;
    }
  }
  __syncthreads();

  // ---- GEMM 2: [16 x 128] @ [128 x 128] + scatter-add ----
#pragma unroll 1
  for (int n0 = 0; n0 < 128; n0 += 16) {
    v8f acc = {};
    const bf16_t* Bb = w2t + (size_t)n0 * 128;
#pragma unroll
    for (int kc = 0; kc < 4; ++kc) {
      v16bf a = load_frag16(&m1[wave][0][kc * 32], 128, lane);
      v16bf b = load_frag16(Bb + kc * 32, 128, lane);
      acc = WMMA_BF16(a, b, acc);
    }
    const int col = n0 + (lane & 15);
    const float bias = b2[col];
#pragma unroll
    for (int r = 0; r < 8; ++r) {
      float v = silu_f(acc[r] + bias);
      atomicAdd(&aggr[(size_t)dsts[wave][rbase + r] * 128 + col], v);
    }
  }
}

// ---------------------------------------------------------------------------
// Node update kernel: per wave, 16 nodes.
//   in = [h | aggr] (16 x 256 bf16): h half via async L2->LDS; aggr half
//   converted f32->bf16 through VGPRs.
//   u1 = silu(in @ w1 + b1); u = u1 @ w2 + b2; h += u (f32 master + bf16 mirror)
// ---------------------------------------------------------------------------
__global__ void __launch_bounds__(128)
node_kernel(float* __restrict__ h, bf16_t* __restrict__ hb,
            const float* __restrict__ aggr,
            const bf16_t* __restrict__ w1t, const float* __restrict__ b1,
            const bf16_t* __restrict__ w2t, const float* __restrict__ b2, int N) {
  __shared__ bf16_t in[4][16][256];   // 32 KB
  __shared__ bf16_t u1[4][16][128];   // 16 KB
  const int wave = threadIdx.x >> 5, lane = threadIdx.x & 31;
  int n0base = (blockIdx.x * 4 + wave) * 16;
  if (n0base + 16 > N) n0base = N - 16;  // tail clamp (N is a multiple of 64)

#pragma unroll 4
  for (int e = 0; e < 16; ++e) {
    const size_t node = n0base + e;
    if (lane < 16) {  // h half: async, 16 lanes x 16B = 256B row
      const bf16_t* grow = hb + node * 128 + lane * 8;
      const unsigned loff = lds_off_u32(&in[wave][e][lane * 8]);
      ASYNC_LD_B128(loff, grow);
    }
    const float4 af = ((const float4*)(aggr + node * 128))[lane];
    bf16_t* q = &in[wave][e][128] + lane * 4;
    q[0] = (bf16_t)af.x; q[1] = (bf16_t)af.y; q[2] = (bf16_t)af.z; q[3] = (bf16_t)af.w;
  }
  WAIT_ASYNCCNT0();
  __syncthreads();

  const int rbase = (lane >> 4) << 3;
#pragma unroll 1
  for (int n0 = 0; n0 < 128; n0 += 16) {
    v8f acc = {};
    const bf16_t* Bb = w1t + (size_t)n0 * 256;
#pragma unroll
    for (int kc = 0; kc < 8; ++kc) {
      v16bf a = load_frag16(&in[wave][0][kc * 32], 256, lane);
      v16bf b = load_frag16(Bb + kc * 32, 256, lane);
      acc = WMMA_BF16(a, b, acc);
    }
    const int col = n0 + (lane & 15);
    const float bias = b1[col];
#pragma unroll
    for (int r = 0; r < 8; ++r) {
      float v = silu_f(acc[r] + bias);
      u1[wave][rbase + r][col] = (bf16_t)v;
    }
  }
  __syncthreads();

#pragma unroll 1
  for (int n0 = 0; n0 < 128; n0 += 16) {
    v8f acc = {};
    const bf16_t* Bb = w2t + (size_t)n0 * 128;
#pragma unroll
    for (int kc = 0; kc < 4; ++kc) {
      v16bf a = load_frag16(&u1[wave][0][kc * 32], 128, lane);
      v16bf b = load_frag16(Bb + kc * 32, 128, lane);
      acc = WMMA_BF16(a, b, acc);
    }
    const int col = n0 + (lane & 15);
    const float bias = b2[col];
#pragma unroll
    for (int r = 0; r < 8; ++r) {
      const size_t idx = (size_t)(n0base + rbase + r) * 128 + col;
      float nv = h[idx] + acc[r] + bias;   // residual, no activation on w2
      h[idx] = nv;
      hb[idx] = (bf16_t)nv;
    }
  }
}

// ---------------------------------------------------------------------------
// Graph mean pooling
// ---------------------------------------------------------------------------
__global__ void pool_accum_kernel(const float* __restrict__ h, const int* __restrict__ batch,
                                  float* __restrict__ sums, float* __restrict__ cnts, int N) {
  int i = blockIdx.x * blockDim.x + threadIdx.x;
  if (i >= N * 128) return;
  int n = i >> 7, c = i & 127;
  int b = batch[n];
  atomicAdd(&sums[(size_t)b * 128 + c], h[i]);
  if (c == 0) atomicAdd(&cnts[b], 1.0f);
}

__global__ void pool_div_kernel(const float* __restrict__ sums, const float* __restrict__ cnts,
                                float* __restrict__ g, bf16_t* __restrict__ gb, int G) {
  int i = blockIdx.x * blockDim.x + threadIdx.x;
  if (i >= G * 128) return;
  float c = fmaxf(cnts[i >> 7], 1.0f);
  float v = sums[i] / c;
  g[i] = v;
  gb[i] = (bf16_t)v;
}

// ---------------------------------------------------------------------------
// Generic bf16 WMMA GEMM: C[M,N] = act(A[M,K] @ Bt[N,K]^T + bias)
// A: bf16 row-major (lda), Bt: bf16 pre-transposed [N][Kpad] (ldb).
// One 16x16 output tile per wave, 4 waves per block.
// act: 0 = none, 1 = relu, 2 = silu.  Optional bf16 mirror Cb.
// ---------------------------------------------------------------------------
__global__ void __launch_bounds__(128)
gemm_bf16_kernel(const bf16_t* __restrict__ A, int lda,
                 const bf16_t* __restrict__ Bt, int ldb,
                 const float* __restrict__ bias,
                 float* __restrict__ C, int ldc,
                 bf16_t* __restrict__ Cb, int ldcb,
                 int Mtiles, int Ntiles, int Kchunks, int act) {
  const int wave = threadIdx.x >> 5, lane = threadIdx.x & 31;
  const int tile = blockIdx.x * 4 + wave;
  if (tile >= Mtiles * Ntiles) return;
  const int mt = tile / Ntiles, nt = tile % Ntiles;
  const bf16_t* Abase = A + (size_t)mt * 16 * lda;
  const bf16_t* Bbase = Bt + (size_t)nt * 16 * ldb;
  v8f acc = {};
  for (int kc = 0; kc < Kchunks; ++kc) {
    __builtin_prefetch(Bbase + (kc + 1) * 32, 0, 1);
    v16bf a = load_frag16(Abase + kc * 32, lda, lane);
    v16bf b = load_frag16(Bbase + kc * 32, ldb, lane);
    acc = WMMA_BF16(a, b, acc);
  }
  const int col = nt * 16 + (lane & 15);
  const int rbase = (lane >> 4) << 3;
  const float bv = bias ? bias[col] : 0.0f;
#pragma unroll
  for (int r = 0; r < 8; ++r) {
    float v = acc[r] + bv;
    if (act == 1) v = fmaxf(v, 0.0f);
    else if (act == 2) v = silu_f(v);
    const int row = mt * 16 + rbase + r;
    C[(size_t)row * ldc + col] = v;
    if (Cb) Cb[(size_t)row * ldcb + col] = (bf16_t)v;
  }
}

// ---------------------------------------------------------------------------
// Attention: G=256, NH=4, HD=32. One thread per (query,head), online softmax.
// Writes bf16 o (input to out-projection GEMM).
// ---------------------------------------------------------------------------
__global__ void attn_kernel(const float* __restrict__ qkv, bf16_t* __restrict__ o, int G) {
  int t = blockIdx.x * blockDim.x + threadIdx.x;
  if (t >= G * 4) return;
  const int q = t >> 2, hh = t & 3;
  const float* qp = qkv + (size_t)q * 384 + hh * 32;
  float m = -1e30f, s = 0.0f;
  float acc[32];
#pragma unroll
  for (int d = 0; d < 32; ++d) acc[d] = 0.0f;
  for (int k = 0; k < G; ++k) {
    const float* kp = qkv + (size_t)k * 384 + 128 + hh * 32;
    float dot = 0.0f;
#pragma unroll
    for (int d = 0; d < 32; ++d) dot += qp[d] * kp[d];
    dot *= 0.17677669529663688f;  // 1/sqrt(32)
    const float nm = fmaxf(m, dot);
    const float corr = __expf(m - nm);
    const float w = __expf(dot - nm);
    s = s * corr + w;
    const float* vp = qkv + (size_t)k * 384 + 256 + hh * 32;
#pragma unroll
    for (int d = 0; d < 32; ++d) acc[d] = acc[d] * corr + w * vp[d];
    m = nm;
  }
  const float inv = 1.0f / s;
#pragma unroll
  for (int d = 0; d < 32; ++d)
    o[(size_t)q * 128 + hh * 32 + d] = (bf16_t)(acc[d] * inv);
}

// ---------------------------------------------------------------------------
// Residual add + LayerNorm over H=128. One block per row, 128 threads.
// ---------------------------------------------------------------------------
__global__ void __launch_bounds__(128)
add_ln_kernel(const float* __restrict__ gin, const float* __restrict__ add,
              const float* __restrict__ gamma, const float* __restrict__ beta,
              float* __restrict__ gout, bf16_t* __restrict__ gb) {
  __shared__ float red[128];
  const int row = blockIdx.x, tdx = threadIdx.x;
  const float x = gin[(size_t)row * 128 + tdx] + add[(size_t)row * 128 + tdx];
  red[tdx] = x;
  __syncthreads();
  for (int o = 64; o > 0; o >>= 1) { if (tdx < o) red[tdx] += red[tdx + o]; __syncthreads(); }
  const float mean = red[0] * (1.0f / 128.0f);
  __syncthreads();
  const float d = x - mean;
  red[tdx] = d * d;
  __syncthreads();
  for (int o = 64; o > 0; o >>= 1) { if (tdx < o) red[tdx] += red[tdx + o]; __syncthreads(); }
  const float var = red[0] * (1.0f / 128.0f);
  const float y = d * rsqrtf(var + 1e-5f) * gamma[tdx] + beta[tdx];
  gout[(size_t)row * 128 + tdx] = y;
  gb[(size_t)row * 128 + tdx] = (bf16_t)y;
}

__global__ void final_kernel(const float* __restrict__ c2, const float* __restrict__ cls_b2,
                             float* __restrict__ out, int G) {
  int i = blockIdx.x * blockDim.x + threadIdx.x;
  if (i >= G) return;
  out[i] = c2[(size_t)i * 16] + cls_b2[0];
}

// ---------------------------------------------------------------------------
// Host side
// ---------------------------------------------------------------------------
static inline size_t alignup(size_t x) { return (x + 255) & ~(size_t)255; }

static void launch_wconv(const float* in, bf16_t* out, int K, int Nsrc, int Npad,
                         int Kpad, hipStream_t s) {
  const int total = Npad * Kpad;
  wconv_kernel<<<(total + 255) / 256, 256, 0, s>>>(in, out, K, Nsrc, Npad, Kpad);
}

extern "C" void kernel_launch(void* const* d_in, const int* in_sizes, int n_in,
                              void* d_out, int out_size, void* d_ws, size_t ws_size,
                              hipStream_t stream) {
  const int H = 128, L = 3, T = 2;
  const int N = in_sizes[29];          // batch vector length
  const int E = in_sizes[28] / 2;      // edge_index [2,E]
  const int G = out_size;              // graphs

  const float* x       = (const float*)d_in[0];
  const float* pos     = (const float*)d_in[1];
  const float* emb_w   = (const float*)d_in[2];
  const float* emb_b   = (const float*)d_in[3];
  const float* edge_w1 = (const float*)d_in[4];
  const float* edge_b1 = (const float*)d_in[5];
  const float* edge_w2 = (const float*)d_in[6];
  const float* edge_b2 = (const float*)d_in[7];
  const float* node_w1 = (const float*)d_in[8];
  const float* node_b1 = (const float*)d_in[9];
  const float* node_w2 = (const float*)d_in[10];
  const float* node_b2 = (const float*)d_in[11];
  const float* qkv_w   = (const float*)d_in[12];
  const float* qkv_b   = (const float*)d_in[13];
  const float* out_w   = (const float*)d_in[14];
  const float* out_b   = (const float*)d_in[15];
  const float* ln1_g   = (const float*)d_in[16];
  const float* ln1_b   = (const float*)d_in[17];
  const float* ln2_g   = (const float*)d_in[18];
  const float* ln2_b   = (const float*)d_in[19];
  const float* ff_w1   = (const float*)d_in[20];
  const float* ff_b1   = (const float*)d_in[21];
  const float* ff_w2   = (const float*)d_in[22];
  const float* ff_b2   = (const float*)d_in[23];
  const float* cls_w1  = (const float*)d_in[24];
  const float* cls_b1  = (const float*)d_in[25];
  const float* cls_w2  = (const float*)d_in[26];
  const float* cls_b2  = (const float*)d_in[27];
  const int*   eidx    = (const int*)d_in[28];
  const int*   src     = eidx;
  const int*   dst     = eidx + E;
  const int*   batch   = (const int*)d_in[29];
  float*       out     = (float*)d_out;

  // ---- carve workspace ----
  char* ws = (char*)d_ws;
  size_t off = 0;
  auto alloc = [&](size_t bytes) -> void* {
    void* p = ws + off; off = alignup(off + bytes); return p;
  };
  float*  h        = (float*)alloc((size_t)N * H * 4);
  bf16_t* hb       = (bf16_t*)alloc((size_t)N * H * 2);
  float*  aggr     = (float*)alloc((size_t)N * H * 4);
  float*  distb    = (float*)alloc((size_t)E * 4);
  bf16_t* edge_w1t = (bf16_t*)alloc((size_t)L * H * 288 * 2);
  bf16_t* edge_w2t = (bf16_t*)alloc((size_t)L * H * 128 * 2);
  bf16_t* node_w1t = (bf16_t*)alloc((size_t)L * H * 256 * 2);
  bf16_t* node_w2t = (bf16_t*)alloc((size_t)L * H * 128 * 2);
  bf16_t* qkv_wt   = (bf16_t*)alloc((size_t)T * 384 * 128 * 2);
  bf16_t* out_wt   = (bf16_t*)alloc((size_t)T * 128 * 128 * 2);
  bf16_t* ff_w1t   = (bf16_t*)alloc((size_t)T * 256 * 128 * 2);
  bf16_t* ff_w2t   = (bf16_t*)alloc((size_t)T * 128 * 256 * 2);
  bf16_t* cls_w1t  = (bf16_t*)alloc((size_t)64 * 128 * 2);
  bf16_t* cls_w2t  = (bf16_t*)alloc((size_t)16 * 64 * 2);
  float*  sums     = (float*)alloc((size_t)G * H * 4);
  float*  cnts     = (float*)alloc((size_t)G * 4);
  float*  g        = (float*)alloc((size_t)G * H * 4);
  bf16_t* gb       = (bf16_t*)alloc((size_t)G * H * 2);
  float*  qkvbuf   = (float*)alloc((size_t)G * 384 * 4);
  bf16_t* o_bf     = (bf16_t*)alloc((size_t)G * 128 * 2);
  float*  oproj    = (float*)alloc((size_t)G * 128 * 4);
  float*  ff1f     = (float*)alloc((size_t)G * 256 * 4);
  bf16_t* ff1b     = (bf16_t*)alloc((size_t)G * 256 * 2);
  float*  ff2      = (float*)alloc((size_t)G * 128 * 4);
  float*  c1f      = (float*)alloc((size_t)G * 64 * 4);
  bf16_t* c1b      = (bf16_t*)alloc((size_t)G * 64 * 2);
  float*  c2       = (float*)alloc((size_t)G * 16 * 4);
  (void)ws_size; (void)n_in;

  // ---- weight conversion (fp32 [K][N] -> bf16 [N][Kpad]) ----
  for (int l = 0; l < L; ++l) {
    launch_wconv(edge_w1 + (size_t)l * 257 * H, edge_w1t + (size_t)l * H * 288, 257, H, H, 288, stream);
    launch_wconv(edge_w2 + (size_t)l * H * H,   edge_w2t + (size_t)l * H * 128, 128, H, H, 128, stream);
    launch_wconv(node_w1 + (size_t)l * 256 * H, node_w1t + (size_t)l * H * 256, 256, H, H, 256, stream);
    launch_wconv(node_w2 + (size_t)l * H * H,   node_w2t + (size_t)l * H * 128, 128, H, H, 128, stream);
  }
  for (int t = 0; t < T; ++t) {
    launch_wconv(qkv_w + (size_t)t * H * 384, qkv_wt + (size_t)t * 384 * 128, 128, 384, 384, 128, stream);
    launch_wconv(out_w + (size_t)t * H * H,   out_wt + (size_t)t * 128 * 128, 128, 128, 128, 128, stream);
    launch_wconv(ff_w1 + (size_t)t * H * 256, ff_w1t + (size_t)t * 256 * 128, 128, 256, 256, 128, stream);
    launch_wconv(ff_w2 + (size_t)t * 256 * H, ff_w2t + (size_t)t * 128 * 256, 256, 128, 128, 256, stream);
  }
  launch_wconv(cls_w1, cls_w1t, 128, 64, 64, 128, stream);
  launch_wconv(cls_w2, cls_w2t, 64, 1, 16, 64, stream);

  // ---- distances + embedding ----
  dist_kernel<<<(E + 255) / 256, 256, 0, stream>>>(pos, src, dst, distb, E);
  embed_kernel<<<((size_t)N * H + 255) / 256, 256, 0, stream>>>(x, emb_w, emb_b, h, hb, N);

  // ---- GNN layers ----
  const int edge_blocks = (E + 63) / 64;
  const int node_blocks = (N + 63) / 64;
  for (int l = 0; l < L; ++l) {
    hipMemsetAsync(aggr, 0, (size_t)N * H * 4, stream);
    edge_kernel<<<edge_blocks, 128, 0, stream>>>(
        hb, distb, src, dst,
        edge_w1t + (size_t)l * H * 288, edge_b1 + (size_t)l * H,
        edge_w2t + (size_t)l * H * 128, edge_b2 + (size_t)l * H, aggr, E);
    node_kernel<<<node_blocks, 128, 0, stream>>>(
        h, hb, aggr,
        node_w1t + (size_t)l * H * 256, node_b1 + (size_t)l * H,
        node_w2t + (size_t)l * H * 128, node_b2 + (size_t)l * H, N);
  }

  // ---- mean pooling ----
  hipMemsetAsync(sums, 0, (size_t)G * H * 4, stream);
  hipMemsetAsync(cnts, 0, (size_t)G * 4, stream);
  pool_accum_kernel<<<((size_t)N * H + 255) / 256, 256, 0, stream>>>(h, batch, sums, cnts, N);
  pool_div_kernel<<<((size_t)G * H + 255) / 256, 256, 0, stream>>>(sums, cnts, g, gb, G);

  // ---- transformer encoder (2 layers, post-norm, relu FFN) ----
  const int Mt = G / 16;  // 16-row tiles
  for (int t = 0; t < T; ++t) {
    gemm_bf16_kernel<<<(Mt * 24 + 3) / 4, 128, 0, stream>>>(
        gb, 128, qkv_wt + (size_t)t * 384 * 128, 128, qkv_b + (size_t)t * 384,
        qkvbuf, 384, (bf16_t*)nullptr, 0, Mt, 24, 4, 0);
    attn_kernel<<<(G * 4 + 127) / 128, 128, 0, stream>>>(qkvbuf, o_bf, G);
    gemm_bf16_kernel<<<(Mt * 8 + 3) / 4, 128, 0, stream>>>(
        o_bf, 128, out_wt + (size_t)t * 128 * 128, 128, out_b + (size_t)t * 128,
        oproj, 128, (bf16_t*)nullptr, 0, Mt, 8, 4, 0);
    add_ln_kernel<<<G, 128, 0, stream>>>(g, oproj, ln1_g + (size_t)t * H, ln1_b + (size_t)t * H, g, gb);
    gemm_bf16_kernel<<<(Mt * 16 + 3) / 4, 128, 0, stream>>>(
        gb, 128, ff_w1t + (size_t)t * 256 * 128, 128, ff_b1 + (size_t)t * 256,
        ff1f, 256, ff1b, 256, Mt, 16, 4, 1 /*relu*/);
    gemm_bf16_kernel<<<(Mt * 8 + 3) / 4, 128, 0, stream>>>(
        ff1b, 256, ff_w2t + (size_t)t * 128 * 256, 256, ff_b2 + (size_t)t * 128,
        ff2, 128, (bf16_t*)nullptr, 0, Mt, 8, 8, 0);
    add_ln_kernel<<<G, 128, 0, stream>>>(g, ff2, ln2_g + (size_t)t * H, ln2_b + (size_t)t * H, g, gb);
  }

  // ---- classifier ----
  gemm_bf16_kernel<<<(Mt * 4 + 3) / 4, 128, 0, stream>>>(
      gb, 128, cls_w1t, 128, cls_b1, c1f, 64, c1b, 64, Mt, 4, 4, 1 /*relu*/);
  gemm_bf16_kernel<<<(Mt * 1 + 3) / 4, 128, 0, stream>>>(
      c1b, 64, cls_w2t, 64, (const float*)nullptr, c2, 16, (bf16_t*)nullptr, 0, Mt, 1, 2, 0);
  final_kernel<<<(G + 255) / 256, 256, 0, stream>>>(c2, cls_b2, out, G);
}
